// EdgePredictor_25314537243118
// MI455X (gfx1250) — matrix operand order repaired and checked by reading, not compiled, not run on hardware
//
#include <hip/hip_runtime.h>

// CDNA5 / gfx1250: wave32, WMMA 16x16x32 f16 -> f32 accumulate.
typedef __attribute__((ext_vector_type(16))) _Float16 v16h;
typedef __attribute__((ext_vector_type(8)))  float    v8f;
typedef __attribute__((ext_vector_type(4)))  unsigned v4u;

union ABFrag { v4u q[2]; v16h h; };   // 32 bytes: 2 x b128 == 16 halves
union DFrag  { v8f v; float f[8]; };

// LDS strides (elements), padded so per-lane fragment addresses spread over
// the 64 x 4B LDS banks instead of aliasing (stride % 64 dwords != 0).
#define XA_S 136   // sXa row stride in halves  (68 dwords,  68 % 64 = 4)
#define WT_S 264   // sWt row stride in halves  (132 dwords, 132 % 64 = 4)
#define HJ_S 132   // sHi/sHj row stride floats (132 % 64 = 4)

// ---------------- Threefry-2x32 (JAX default PRNG), key = (0, 42) -----------
__device__ __forceinline__ unsigned rotl32(unsigned x, unsigned n) {
  return (x << n) | (x >> (32u - n));
}

__device__ __forceinline__ void threefry2x32_k42(unsigned x0, unsigned x1,
                                                 unsigned& o0, unsigned& o1) {
  const unsigned k0 = 0u, k1 = 42u;
  const unsigned k2 = k0 ^ k1 ^ 0x1BD11BDAu;
  unsigned a = x0 + k0, b = x1 + k1;
#define TF_R(r) { a += b; b = rotl32(b, (r)); b ^= a; }
  TF_R(13) TF_R(15) TF_R(26) TF_R(6)
  a += k1; b += k2 + 1u;
  TF_R(17) TF_R(29) TF_R(16) TF_R(24)
  a += k2; b += k0 + 2u;
  TF_R(13) TF_R(15) TF_R(26) TF_R(6)
  a += k0; b += k1 + 3u;
  TF_R(17) TF_R(29) TF_R(16) TF_R(24)
  a += k1; b += k2 + 4u;
  TF_R(13) TF_R(15) TF_R(26) TF_R(6)
  a += k2; b += k0 + 5u;
#undef TF_R
  o0 = a; o1 = b;
}

// Gumbel sample matching jax.random.gumbel(key(42), (64,64,64,2), f32) at
// flat element index m in [0, 524288). JAX splits iota(524288) in half and
// runs Threefry on pairs (p, p+262144); result is concat(x0_all, x1_all).
__device__ __forceinline__ float gumbel_at(unsigned m) {
  const unsigned H = 262144u;
  unsigned o0, o1, bits;
  if (m < H) { threefry2x32_k42(m, m + H, o0, o1); bits = o0; }
  else       { threefry2x32_k42(m - H, m, o0, o1); bits = o1; }
  unsigned fb = (bits >> 9) | 0x3F800000u;            // [1, 2)
  float f = __uint_as_float(fb) - 1.0f;               // [0, 1)
  const float tiny = 1.17549435e-38f;
  float u = fmaxf(tiny, f + tiny);                    // JAX uniform(minval=tiny)
  return -logf(-logf(u));
}

// ---------------- Fused edge-predictor kernel: one block per batch b --------
__global__ __launch_bounds__(256)
void edge_predictor_kernel(const float* __restrict__ xa,   // (64,64,128)
                           const float* __restrict__ W1,   // (256,128)
                           const float* __restrict__ b1,   // (128)
                           const float* __restrict__ W2,   // (128,2)
                           const float* __restrict__ b2,   // (2)
                           float* __restrict__ out) {      // (64,64,64)
  const int b    = blockIdx.x;
  const int tid  = threadIdx.x;
  const int lane = tid & 31;
  const int wave = tid >> 5;

  extern __shared__ char smem[];
  float*    sHi  = (float*)smem;                   // 64 x HJ_S f32 (hi + b1)
  float*    sHj  = sHi + 64 * HJ_S;                // 64 x HJ_S f32
  _Float16* sXa  = (_Float16*)(sHj + 64 * HJ_S);   // 64 x XA_S f16 (row-major)
  _Float16* sWt  = sXa + 64 * XA_S;                // 128 x WT_S f16 (W1^T: [h][k])
  float*    sW2d = (float*)(sWt + 128 * WT_S);     // 128 f32 (W2[:,1]-W2[:,0])
  float*    sB1  = sW2d + 128;                     // 128 f32

  __builtin_prefetch(W1 + tid * 128, 0, 0);        // global_prefetch_b8

  // ---- Stage 0: stage xa[b] (row-major) and W1 (transposed) as f16 ---------
  const float* xab = xa + b * 64 * 128;
  for (int i = tid; i < 64 * 128; i += 256) {
    sXa[(i >> 7) * XA_S + (i & 127)] = (_Float16)xab[i];
  }
  for (int i = tid; i < 256 * 128; i += 256) {
    const int k = i >> 7, h = i & 127;             // W1 is (k, h) row-major
    sWt[h * WT_S + k] = (_Float16)W1[i];           // transpose into [h][k]
  }
  if (tid < 128) {
    sB1[tid]  = b1[tid];
    sW2d[tid] = W2[tid * 2 + 1] - W2[tid * 2 + 0];
  }
  __syncthreads();

  // ---- Stage 1: Y = xa[b](64x128) @ [W1i | W1j](128x256) via WMMA ----------
  // Wave w owns M-tile (w>>1) and the hi half (w even) or hj half (w odd):
  // 8 N-tiles x 4 K-steps per wave. The 4 A-fragments are shared by all 8
  // N-tiles -> hoist them into registers; inner loop is 2x ds_load_b128 (B)
  // per v_wmma. Epilogue is branchless (bias/dst selected, not branched).
  const int hs = lane >> 4;   // lane half-select
  const int lr = lane & 15;
  const int mt     = wave >> 1;                  // 0..3
  const int isHj   = wave & 1;                   // 0 -> hi, 1 -> hj
  const int kofs   = isHj ? 128 : 0;             // W1j rows start at k=128
  float*    dst    = isHj ? sHj : sHi;
  const int r      = (mt << 4) + lr;             // A row

  ABFrag afr[4];
#pragma unroll
  for (int kk = 0; kk < 4; ++kk) {
    // A 16x32 f16 layout: lane half s holds K runs {0..7}+8s and {16..23}+8s
    const _Float16* abase = &sXa[r * XA_S + kk * 32 + 8 * hs];
    afr[kk].q[0] = *(const v4u*)(const void*)(abase);
    afr[kk].q[1] = *(const v4u*)(const void*)(abase + 16);
  }

  for (int t = 0; t < 8; ++t) {
    const int c = (t << 4) + lr;                 // output column within H

    DFrag acc;
#pragma unroll
    for (int v = 0; v < 8; ++v) acc.f[v] = 0.0f;

#pragma unroll
    for (int kk = 0; kk < 4; ++kk) {
      // B 32x16 f16 layout: lane = column c, half s holds K runs {0..7}+8s and {16..23}+8s
      // -> two 8-half runs along k in transposed sWt -> 2 x ds_load_b128.
      ABFrag bf;
      const _Float16* bbase = &sWt[c * WT_S + kofs + kk * 32 + 8 * hs];
      bf.q[0] = *(const v4u*)(const void*)(bbase);
      bf.q[1] = *(const v4u*)(const void*)(bbase + 16);

      acc.v = __builtin_amdgcn_wmma_f32_16x16x32_f16(
          false, afr[kk].h, false, bf.h, (short)0, acc.v, false, false);
    }

    // Branchless epilogue: fold b1 into hi rows only (select, no exec branch).
    const float bv   = sB1[c];
    const float bias = isHj ? 0.0f : bv;
    // D 16x16 f32 layout: lane = column, VGPR v = row v + 8*(lane>=16)
#pragma unroll
    for (int v = 0; v < 8; ++v) {
      const int row = (mt << 4) + v + 8 * hs;
      dst[row * HJ_S + c] = acc.f[v] + bias;
    }
  }
  __syncthreads();

  // ---- Stage 2: edge[b,i,j] = sigmoid(dot(relu(hi[i]+hj[j]), W2d)
  //                                     + b2d + g1 - g0) ---------------------
  // i distinct per lane (bank-spread hi reads); j-strip uniform per wave
  // (hj reads are same-address LDS broadcasts).
  const float b2d = b2[1] - b2[0];
  const int i  = tid & 63;          // 0..63, distinct within a wave
  const int jg = (tid >> 6) << 4;   // 16-wide j strip, wave-uniform

  float acc2[16];
#pragma unroll
  for (int j = 0; j < 16; ++j) acc2[j] = 0.0f;

  for (int h = 0; h < 128; h += 4) {
    const float4 hi4 = *(const float4*)(const void*)(&sHi[i * HJ_S + h]);
    const float4 w4  = *(const float4*)(const void*)(&sW2d[h]);
#pragma unroll
    for (int j = 0; j < 16; ++j) {
      const float4 hj4 = *(const float4*)(const void*)(&sHj[(jg + j) * HJ_S + h]);
      float s = acc2[j];
      s = fmaf(fmaxf(hi4.x + hj4.x, 0.0f), w4.x, s);
      s = fmaf(fmaxf(hi4.y + hj4.y, 0.0f), w4.y, s);
      s = fmaf(fmaxf(hi4.z + hj4.z, 0.0f), w4.z, s);
      s = fmaf(fmaxf(hi4.w + hj4.w, 0.0f), w4.w, s);
      acc2[j] = s;
    }
  }

  const unsigned base = ((unsigned)b * 64u + (unsigned)i) * 64u + (unsigned)jg;
#pragma unroll
  for (int j = 0; j < 16; ++j) {
    const unsigned n = base + (unsigned)j;
    const float g0 = gumbel_at(2u * n);
    const float g1 = gumbel_at(2u * n + 1u);
    const float z  = acc2[j] + b2d + (g1 - g0);
    out[n] = 1.0f / (1.0f + expf(-z));   // softmax over 2 == sigmoid(z1-z0)
  }
}

extern "C" void kernel_launch(void* const* d_in, const int* in_sizes, int n_in,
                              void* d_out, int out_size, void* d_ws, size_t ws_size,
                              hipStream_t stream) {
  (void)in_sizes; (void)n_in; (void)out_size; (void)d_ws; (void)ws_size;
  const float* xa = (const float*)d_in[0];
  const float* W1 = (const float*)d_in[1];
  const float* b1 = (const float*)d_in[2];
  const float* W2 = (const float*)d_in[3];
  const float* b2 = (const float*)d_in[4];
  float* out = (float*)d_out;

  const size_t shmem = (size_t)(64 * HJ_S) * 4 * 2   // sHi, sHj      67584 B
                     + (size_t)(64 * XA_S) * 2       // sXa f16       17408 B
                     + (size_t)(128 * WT_S) * 2      // sWt f16       67584 B
                     + 128 * 4 * 2;                  // sW2d, sB1      1024 B
  edge_predictor_kernel<<<64, 256, shmem, stream>>>(xa, W1, b1, W2, b2, out);
}
